// CustomLSTMCell_6820408066632
// MI455X (gfx1250) — matrix operand order repaired and checked by reading, not compiled, not run on hardware
//
#include <hip/hip_runtime.h>

typedef __attribute__((ext_vector_type(16))) __bf16   v16bf;
typedef __attribute__((ext_vector_type(8)))  float    v8f;
typedef __attribute__((ext_vector_type(4)))  unsigned int u32x4;

static __device__ __forceinline__ unsigned short f2bf(float f) {
    unsigned int u = __float_as_uint(f);
    unsigned int r = u + 0x7fffu + ((u >> 16) & 1u);   // round-to-nearest-even
    return (unsigned short)(r >> 16);
}

static __device__ __forceinline__ float fsigmoid(float x) {
    return 1.0f / (1.0f + __expf(-x));
}

static __device__ __forceinline__ float ftanh(float x) {
    float ax = fabsf(x);
    float t  = 1.0f - 2.0f / (__expf(2.0f * ax) + 1.0f);
    return copysignf(t, x);
}

// ---------------------------------------------------------------------------
// Kernel 1: A = bf16(concat(inputs, h_tm1)), row-major 8192 x 2048.
// ---------------------------------------------------------------------------
__global__ __launch_bounds__(256) void pack_A_kernel(
    const float* __restrict__ x, const float* __restrict__ h,
    unsigned short* __restrict__ A)
{
    size_t e   = ((size_t)blockIdx.x * 256 + threadIdx.x) * 8;
    size_t row = e >> 11;                 // / 2048
    unsigned col = (unsigned)e & 2047u;
    const float* src = (col < 1024u) ? (x + row * 1024 + col)
                                     : (h + row * 1024 + (col - 1024u));
    float4 v0 = *(const float4*)(src);
    float4 v1 = *(const float4*)(src + 4);
    u32x4 o;
    o.x = (unsigned)f2bf(v0.x) | ((unsigned)f2bf(v0.y) << 16);
    o.y = (unsigned)f2bf(v0.z) | ((unsigned)f2bf(v0.w) << 16);
    o.z = (unsigned)f2bf(v1.x) | ((unsigned)f2bf(v1.y) << 16);
    o.w = (unsigned)f2bf(v1.z) | ((unsigned)f2bf(v1.w) << 16);
    *(u32x4*)(A + e) = o;
}

// ---------------------------------------------------------------------------
// Kernel 2: Wt[n][k] = bf16(W[k][n]) for the fused weight matrix.
// Wt is 4096 rows (gate*1024 + n) x 2048 cols (k: 0..1023 = W, 1024..2047 = U).
// ---------------------------------------------------------------------------
__global__ __launch_bounds__(256) void transpose_W_kernel(
    const float* __restrict__ W_i, const float* __restrict__ U_i,
    const float* __restrict__ W_f, const float* __restrict__ U_f,
    const float* __restrict__ W_c, const float* __restrict__ U_c,
    const float* __restrict__ W_o, const float* __restrict__ U_o,
    unsigned short* __restrict__ Wt)
{
    __shared__ float tile[32][33];
    const int z = blockIdx.z;
    const float* src =
        (z == 0) ? W_i : (z == 1) ? U_i :
        (z == 2) ? W_f : (z == 3) ? U_f :
        (z == 4) ? W_c : (z == 5) ? U_c :
        (z == 6) ? W_o : U_o;
    const int rowbase = (z >> 1) * 1024;   // gate offset among Wt rows (n)
    const int colbase = (z & 1) * 1024;    // W vs U offset among Wt cols (k)
    const int tx = threadIdx.x & 31;
    const int ty = threadIdx.x >> 5;
    const int n0 = blockIdx.x * 32;        // source column tile
    const int k0 = blockIdx.y * 32;        // source row tile
#pragma unroll
    for (int i = 0; i < 4; ++i) {
        int k = k0 + ty + i * 8;
        tile[ty + i * 8][tx] = src[(size_t)k * 1024 + n0 + tx];
    }
    __syncthreads();
#pragma unroll
    for (int i = 0; i < 4; ++i) {
        int n = n0 + ty + i * 8;
        Wt[(size_t)(rowbase + n) * 2048 + colbase + k0 + tx] =
            f2bf(tile[tx][ty + i * 8]);
    }
}

// ---------------------------------------------------------------------------
// Kernel 3: fused GEMM + LSTM gates, software-pipelined.
// Block: 256 threads = 8 waves arranged 4(M) x 2(N).
// Block tile: 128 rows x 32 cols-per-gate. Wave: 2 M-frags x 4 gates.
// K-loop in steps of 32 with ping-pong fragment buffers; addresses are
// (scalar base = matrix + k) + (fixed 32-bit per-lane offset) so the loop
// carries only scalar pointer arithmetic (GVS/saddr-form loads).
// ---------------------------------------------------------------------------
union Frag { u32x4 u[2]; v16bf v; };

__global__ __launch_bounds__(256) void lstm_gemm_kernel(
    const unsigned short* __restrict__ A,   // 8192 x 2048 bf16, row-major
    const unsigned short* __restrict__ Wt,  // 4096 x 2048 bf16, row-major (N x K)
    const float* __restrict__ bI, const float* __restrict__ bF,
    const float* __restrict__ bC, const float* __restrict__ bO,
    const float* __restrict__ c_prev,
    float* __restrict__ h_out, float* __restrict__ c_out)
{
    constexpr int K  = 2048;
    constexpr int Ug = 1024;
    const int lane = threadIdx.x & 31;
    const int wave = threadIdx.x >> 5;
    const int wm   = wave & 3;
    const int wn   = wave >> 2;
    const int row0 = blockIdx.x * 128 + wm * 32;  // wave rows: row0..row0+31
    const int col0 = blockIdx.y * 32 + wn * 16;   // wave cols within a gate

    // Loop-invariant per-lane element offsets (32-bit, divergent).
    // A fragment (16-bit A 16x32): lane<16 -> row=lane, K {0..7,16..23};
    // lane>=16 -> row=lane-16, K {8..15,24..31}.
    const unsigned aoff0 = (unsigned)(row0 + (lane & 15)) * K + (unsigned)((lane >> 4) * 8);
    const unsigned aoff1 = aoff0 + 16u * K;
    // B fragment (dense 16-bit B 32x16): lane = column; lanes 0-15 -> K {0..7,16..23},
    // lanes 16-31 -> K {8..15,24..31} (matches A).
    unsigned boff[4];
#pragma unroll
    for (int g = 0; g < 4; ++g)
        boff[g] = (unsigned)(g * 1024 + col0 + (lane & 15)) * K
                + (unsigned)((lane >> 4) * 8);

    v8f acc[2][4] = {};
    Frag a0[2], a1[2], b[2][4];

    // Prologue: fragments for k = 0.
    {
        const unsigned short* __restrict__ Ak = A;
        const unsigned short* __restrict__ Wk = Wt;
        a0[0].u[0] = *(const u32x4*)(Ak + aoff0);
        a0[0].u[1] = *(const u32x4*)(Ak + aoff0 + 16);
        a1[0].u[0] = *(const u32x4*)(Ak + aoff1);
        a1[0].u[1] = *(const u32x4*)(Ak + aoff1 + 16);
#pragma unroll
        for (int g = 0; g < 4; ++g) {
            b[0][g].u[0] = *(const u32x4*)(Wk + boff[g]);
            b[0][g].u[1] = *(const u32x4*)(Wk + boff[g] + 16);
        }
    }

#pragma unroll 4
    for (int k = 0; k < K; k += 32) {
        const int cur = (k >> 5) & 1;
        const int nxt = cur ^ 1;
        // Prefetch k-step n+1 (wraps to 0 on the last step; harmless re-load).
        const int kn = (k + 32) & (K - 1);
        const unsigned short* __restrict__ Ak = A + kn;   // uniform (SGPR) base
        const unsigned short* __restrict__ Wk = Wt + kn;  // uniform (SGPR) base
        a0[nxt].u[0] = *(const u32x4*)(Ak + aoff0);
        a0[nxt].u[1] = *(const u32x4*)(Ak + aoff0 + 16);
        a1[nxt].u[0] = *(const u32x4*)(Ak + aoff1);
        a1[nxt].u[1] = *(const u32x4*)(Ak + aoff1 + 16);
#pragma unroll
        for (int g = 0; g < 4; ++g) {
            b[nxt][g].u[0] = *(const u32x4*)(Wk + boff[g]);
            b[nxt][g].u[1] = *(const u32x4*)(Wk + boff[g] + 16);
        }
        // Consume k-step n (its loads were issued one iteration ago).
#pragma unroll
        for (int g = 0; g < 4; ++g) {
            acc[0][g] = __builtin_amdgcn_wmma_f32_16x16x32_bf16(
                false, a0[cur].v, false, b[cur][g].v, (short)0, acc[0][g], false, false);
            acc[1][g] = __builtin_amdgcn_wmma_f32_16x16x32_bf16(
                false, a1[cur].v, false, b[cur][g].v, (short)0, acc[1][g], false, false);
        }
    }

    // Epilogue. C/D layout: VGPR r, lanes 0-15 -> (M=r, N=lane);
    // lanes 16-31 -> (M=r+8, N=lane-16).
    const int colg = col0 + (lane & 15);
    const float biV = bI[colg], bfV = bF[colg], bcV = bC[colg], boV = bO[colg];
    const int rext = (lane >> 4) * 8;
#pragma unroll
    for (int m = 0; m < 2; ++m) {
#pragma unroll
        for (int r = 0; r < 8; ++r) {
            const int row = row0 + m * 16 + rext + r;
            const size_t idx = (size_t)row * Ug + colg;
            float zi = acc[m][0][r] + biV;
            float zf = acc[m][1][r] + bfV;
            float zc = acc[m][2][r] + bcV;
            float zo = acc[m][3][r] + boV;
            float ig = fsigmoid(zi);
            float fg = fsigmoid(zf);
            float og = fsigmoid(zo);
            float cc = ftanh(zc);
            float cp = c_prev[idx];
            float cn = fg * cp + ig * cc;
            h_out[idx] = og * ftanh(cn);
            c_out[idx] = cn;
        }
    }
}

// ---------------------------------------------------------------------------
extern "C" void kernel_launch(void* const* d_in, const int* in_sizes, int n_in,
                              void* d_out, int out_size, void* d_ws, size_t ws_size,
                              hipStream_t stream)
{
    (void)in_sizes; (void)n_in; (void)out_size; (void)ws_size;
    const float* x   = (const float*)d_in[0];
    const float* h   = (const float*)d_in[1];
    const float* c   = (const float*)d_in[2];
    const float* W_i = (const float*)d_in[3];
    const float* U_i = (const float*)d_in[4];
    const float* b_i = (const float*)d_in[5];
    const float* W_f = (const float*)d_in[6];
    const float* U_f = (const float*)d_in[7];
    const float* b_f = (const float*)d_in[8];
    const float* W_c = (const float*)d_in[9];
    const float* U_c = (const float*)d_in[10];
    const float* b_c = (const float*)d_in[11];
    const float* W_o = (const float*)d_in[12];
    const float* U_o = (const float*)d_in[13];
    const float* b_o = (const float*)d_in[14];

    float* h_out = (float*)d_out;
    float* c_out = h_out + (size_t)8192 * 1024;

    // Workspace: A_bf16 (8192x2048 = 32MB) then Wt_bf16 (4096x2048 = 16MB).
    unsigned short* A_bf = (unsigned short*)d_ws;
    unsigned short* Wt   = A_bf + (size_t)8192 * 2048;

    pack_A_kernel<<<8192, 256, 0, stream>>>(x, h, A_bf);
    transpose_W_kernel<<<dim3(32, 32, 8), 256, 0, stream>>>(
        W_i, U_i, W_f, U_f, W_c, U_c, W_o, U_o, Wt);
    lstm_gemm_kernel<<<dim3(8192 / 128, 1024 / 32), 256, 0, stream>>>(
        A_bf, Wt, b_i, b_f, b_c, b_o, c, h_out, c_out);
}
